// BiGAT_1855425872579
// MI455X (gfx1250) — matrix operand order repaired and checked by reading, not compiled, not run on hardware
//
#include <hip/hip_runtime.h>
#include <hip/hip_bf16.h>
#include <math.h>

typedef __bf16 bf16_t;
typedef __attribute__((ext_vector_type(16))) __bf16 v16bf;
typedef __attribute__((ext_vector_type(8)))  __bf16 v8bf;
typedef __attribute__((ext_vector_type(8)))  float  v8f;

#define NODES 20000
#define NEG_SLOPE 0.2f
#define GEMM_WAVES 4

// ---------------- utility kernels ----------------

__global__ void k_f32_to_bf16(const float* __restrict__ in, bf16_t* __restrict__ out, int n) {
    int i = blockIdx.x * blockDim.x + threadIdx.x;
    if (i < n) out[i] = (bf16_t)in[i];
}

__global__ void k_fill_f32(float* __restrict__ p, float v, int n) {
    int i = blockIdx.x * blockDim.x + threadIdx.x;
    if (i < n) p[i] = v;
}

__global__ void k_fill_bits(unsigned* __restrict__ p, unsigned bits, int n) {
    int i = blockIdx.x * blockDim.x + threadIdx.x;
    if (i < n) p[i] = bits;
}

// ---------------- B-matrix pack: fp32 W[K,Nout] -> bf16 WMMA B-fragments ----------------
// B layout (contiguous k, matches A; not ISA table order): col = lane&15; lanes 0-15 hold
// K = kb+0..15, lanes 16-31 hold K = kb+16..31, 16 contiguous bf16 per lane.
// Linear index: (((ntile * K/32 + kb32) * 32) + lane) * 16 + i
// Zero-pads columns >= Nout so the GEMM needs no bounds checks in the K loop.

__global__ void k_pack_B(const float* __restrict__ W, bf16_t* __restrict__ Wpk,
                         int K, int Nout, int NoutPad) {
    int idx = blockIdx.x * blockDim.x + threadIdx.x;
    int total = NoutPad * K;            // total packed elements
    if (idx >= total) return;
    int i    = idx & 15;                // element within lane
    int lane = (idx >> 4) & 31;
    int frag = idx >> 9;                // ntile * (K/32) + kb32
    int nk   = K >> 5;
    int ntile = frag / nk;
    int kb32  = frag - ntile * nk;
    int col = ntile * 16 + (lane & 15);
    int k   = kb32 * 32 + (lane >> 4) * 16 + i;
    float v = (col < Nout) ? W[(size_t)k * Nout + col] : 0.f;
    Wpk[idx] = (bf16_t)v;
}

// ---------------- WMMA bf16 GEMM:  C[M,Nout] = A[M,K] * Bpk ----------------
// 4 waves per block, one 16x16 output tile per wave. K multiple of 32, M multiple of 16.
// A-fragment (16-bit A 16x32): lanes 0-15 row M=lane, K = kb+0..15;
// lanes 16-31 K = kb+16..31. Both halves = 16B contiguous loads.
// B-fragment comes pre-packed: one contiguous 32B chunk per lane per K-step.
// D: VGPR r -> row tm + r + 8*(lane>=16), col = tn + (lane&15).

__global__ void k_gemm_bf16_wmma(const bf16_t* __restrict__ A, int lda,
                                 const bf16_t* __restrict__ Bpk,
                                 float* __restrict__ C,
                                 int K, int Nout, int ldc, int ntiles) {
    int lane = threadIdx.x & 31;
    int wave = threadIdx.x >> 5;
    int tm = blockIdx.x * 16;
    int wt = blockIdx.y * GEMM_WAVES + wave;     // N-tile index for this wave
    if (wt >= ntiles) return;                    // wave-uniform; EXEC stays all-1s
    int tn = wt * 16;
    int r15 = lane & 15;
    int hi  = lane >> 4;

    v8f acc = {0.f, 0.f, 0.f, 0.f, 0.f, 0.f, 0.f, 0.f};

    const bf16_t* arow = A + (size_t)(tm + r15) * lda + hi * 16;
    int nk = K >> 5;
    const bf16_t* bp = Bpk + ((size_t)(wt * nk) * 32 + lane) * 16;

    for (int kb = 0; kb < K; kb += 32) {
        v8bf alo = *(const v8bf*)(arow + kb);        // K = kb + hi*16 + 0..7
        v8bf ahi = *(const v8bf*)(arow + kb + 8);    // K = kb + hi*16 + 8..15
        v16bf b  = *(const v16bf*)bp;                // pre-packed fragment (32B)
        bp += 32 * 16;
        __builtin_prefetch(arow + kb + 64, 0, 1);    // speculative A-stream prefetch

        v16bf a;
#pragma unroll
        for (int i = 0; i < 8; ++i) { a[i] = alo[i]; a[8 + i] = ahi[i]; }

        acc = __builtin_amdgcn_wmma_f32_16x16x32_bf16(
            /*neg_a=*/false, a, /*neg_b=*/false, b,
            /*c_mod=*/(short)0, acc, /*reuse_a=*/false, /*reuse_b=*/false);
    }

    int col = tn + r15;
    if (col < Nout) {
#pragma unroll
        for (int r = 0; r < 8; ++r)
            C[(size_t)(tm + r + hi * 8) * ldc + col] = acc[r];
    }
}

// ---------------- attention kernels ----------------

// alpha_src[n,h] = sum_c h[n,h,c]*a_src[h,c]; likewise alpha_dst
__global__ void k_alpha(const float* __restrict__ h,
                        const float* __restrict__ avs, const float* __restrict__ avd,
                        float* __restrict__ asrc, float* __restrict__ adst,
                        int N, int Hh, int C) {
    int idx = blockIdx.x * blockDim.x + threadIdx.x;
    if (idx >= N * Hh) return;
    int n = idx / Hh, hd = idx - n * Hh;
    const float* hp = h + (size_t)n * Hh * C + (size_t)hd * C;
    const float* as = avs + hd * C;
    const float* ad = avd + hd * C;
    float s = 0.f, d = 0.f;
    for (int c = 0; c < C; ++c) { float v = hp[c]; s += v * as[c]; d += v * ad[c]; }
    asrc[idx] = s; adst[idx] = d;
}

__device__ __forceinline__ void edge_nodes(const int* __restrict__ src,
                                           const int* __restrict__ dst,
                                           int Eh, int e, int& s, int& d) {
    if (e < Eh) { s = src[e]; d = dst[e]; }
    else        { s = e - Eh; d = s; }     // self loops appended
}

// e = leaky_relu(asrc[src]+adst[dst]); segment atomic max over dst
__global__ void k_edge_max(const int* __restrict__ src, const int* __restrict__ dst,
                           int Eh, int Etot,
                           const float* __restrict__ asrc, const float* __restrict__ adst,
                           float* __restrict__ ee, float* __restrict__ emax, int Hh) {
    int idx = blockIdx.x * blockDim.x + threadIdx.x;
    if (idx >= Etot * Hh) return;
    int e = idx / Hh, hd = idx - e * Hh;
    int s, d; edge_nodes(src, dst, Eh, e, s, d);
    float v = asrc[s * Hh + hd] + adst[d * Hh + hd];
    v = (v > 0.f) ? v : NEG_SLOPE * v;
    ee[idx] = v;
    float* addr = &emax[d * Hh + hd];
    // IEEE float atomic-max via monotone integer reinterpretation
    if (v >= 0.f) atomicMax((int*)addr, __float_as_int(v));
    else          atomicMin((unsigned int*)addr, __float_as_uint(v));
}

// ee = exp(ee - emax[dst]); segment atomic sum over dst
__global__ void k_edge_exp(const int* __restrict__ src, const int* __restrict__ dst,
                           int Eh, int Etot,
                           float* __restrict__ ee, const float* __restrict__ emax,
                           float* __restrict__ esum, int Hh) {
    int idx = blockIdx.x * blockDim.x + threadIdx.x;
    if (idx >= Etot * Hh) return;
    int e = idx / Hh, hd = idx - e * Hh;
    int s, d; edge_nodes(src, dst, Eh, e, s, d);
    float v = expf(ee[idx] - emax[d * Hh + hd]);
    ee[idx] = v;
    atomicAdd(&esum[d * Hh + hd], v);
}

// out[dst] += (ee/esum[dst]) * h[src] ; one block per edge, thread per channel
__global__ void k_scatter(const int* __restrict__ src, const int* __restrict__ dst,
                          int Eh, int Etot,
                          const float* __restrict__ ee, const float* __restrict__ esum,
                          const float* __restrict__ h, float* __restrict__ acc,
                          int Hh, int C) {
    int e = blockIdx.x;
    int ch = threadIdx.x;
    int HC = Hh * C;
    if (ch >= HC) return;
    int s, d; edge_nodes(src, dst, Eh, e, s, d);
    int hd = ch / C;
    float al = ee[e * Hh + hd] / (esum[d * Hh + hd] + 1e-16f);
    atomicAdd(&acc[(size_t)d * HC + ch], al * h[(size_t)s * HC + ch]);
}

// xnext[n, coloff+ch] = bf16(elu(acc[n,ch] + bias[ch]))
__global__ void k_finalize_elu(const float* __restrict__ acc, const float* __restrict__ bias,
                               bf16_t* __restrict__ xnext, int ldx, int coloff,
                               int HC, int N) {
    int i = blockIdx.x * blockDim.x + threadIdx.x;
    if (i >= N * HC) return;
    int n = i / HC, ch = i - n * HC;
    float v = acc[i] + bias[ch];
    v = (v > 0.f) ? v : expm1f(v);
    xnext[(size_t)n * ldx + coloff + ch] = (bf16_t)v;
}

// out[n,:] = softmax(acc[n,:] + bias)
__global__ void k_last_softmax(const float* __restrict__ acc, const float* __restrict__ bias,
                               float* __restrict__ out, int N, int C) {
    int n = blockIdx.x * blockDim.x + threadIdx.x;
    if (n >= N) return;
    float buf[40];
    float mx = -3.0e38f;
    for (int c = 0; c < C; ++c) {
        float v = acc[(size_t)n * C + c] + bias[c];
        buf[c] = v;
        mx = fmaxf(mx, v);
    }
    float s = 0.f;
    for (int c = 0; c < C; ++c) { float e = expf(buf[c] - mx); buf[c] = e; s += e; }
    float inv = 1.f / s;
    for (int c = 0; c < C; ++c) out[(size_t)n * C + c] = buf[c] * inv;
}

// ---------------- host orchestration ----------------

static void run_gat_conv(hipStream_t stream,
                         const bf16_t* xb, int lda, int K,
                         const float* W, const float* a_src, const float* a_dst, const float* bias,
                         const int* src, const int* dst, int Eh,
                         int Hh, int C,
                         bf16_t* Wpk, float* h, float* acc,
                         float* asrc, float* adst, float* emax, float* esum, float* ee,
                         bf16_t* xnext, int ldx, int coloff) {
    const int HC = Hh * C;
    const int Nout = HC;
    const int NoutPad = ((Nout + 15) / 16) * 16;
    const int ntiles = NoutPad / 16;

    int pn = NoutPad * K;
    k_pack_B<<<(pn + 255) / 256, 256, 0, stream>>>(W, Wpk, K, Nout, NoutPad);

    dim3 g(NODES / 16, (ntiles + GEMM_WAVES - 1) / GEMM_WAVES);
    k_gemm_bf16_wmma<<<g, GEMM_WAVES * 32, 0, stream>>>(xb, lda, Wpk, h, K, Nout, Nout, ntiles);

    int an = NODES * Hh;
    k_alpha<<<(an + 255) / 256, 256, 0, stream>>>(h, a_src, a_dst, asrc, adst, NODES, Hh, C);
    k_fill_bits<<<(an + 255) / 256, 256, 0, stream>>>((unsigned*)emax, 0xFF800000u, an); // -inf
    k_fill_f32 <<<(an + 255) / 256, 256, 0, stream>>>(esum, 0.f, an);

    int Etot = Eh + NODES;
    int en = Etot * Hh;
    k_edge_max<<<(en + 255) / 256, 256, 0, stream>>>(src, dst, Eh, Etot, asrc, adst, ee, emax, Hh);
    k_edge_exp<<<(en + 255) / 256, 256, 0, stream>>>(src, dst, Eh, Etot, ee, emax, esum, Hh);

    int accn = NODES * HC;
    k_fill_f32<<<(accn + 255) / 256, 256, 0, stream>>>(acc, 0.f, accn);

    int blk = ((HC + 63) / 64) * 64;   // 256 for H=4,C=64; 64 for H=1,C=40
    k_scatter<<<Etot, blk, 0, stream>>>(src, dst, Eh, Etot, ee, esum, h, acc, Hh, C);

    if (xnext)
        k_finalize_elu<<<(accn + 255) / 256, 256, 0, stream>>>(acc, bias, xnext, ldx, coloff, HC, NODES);
}

extern "C" void kernel_launch(void* const* d_in, const int* in_sizes, int n_in,
                              void* d_out, int out_size, void* d_ws, size_t ws_size,
                              hipStream_t stream) {
    (void)in_sizes; (void)n_in; (void)out_size; (void)ws_size;

    const float* x          = (const float*)d_in[0];
    const int*   edge_index = (const int*)d_in[1];   // [2, 640000]
    const int*   st         = (const int*)d_in[2];   // [2, 320000]
    const int*   ts         = (const int*)d_in[3];   // [2, 320000]

    const float* W_st0 = (const float*)d_in[4];
    const float* as_st0 = (const float*)d_in[5];
    const float* ad_st0 = (const float*)d_in[6];
    const float* b_st0  = (const float*)d_in[7];
    const float* W_ts0 = (const float*)d_in[8];
    const float* as_ts0 = (const float*)d_in[9];
    const float* ad_ts0 = (const float*)d_in[10];
    const float* b_ts0  = (const float*)d_in[11];
    const float* W_st1 = (const float*)d_in[12];
    const float* as_st1 = (const float*)d_in[13];
    const float* ad_st1 = (const float*)d_in[14];
    const float* b_st1  = (const float*)d_in[15];
    const float* W_ts1 = (const float*)d_in[16];
    const float* as_ts1 = (const float*)d_in[17];
    const float* ad_ts1 = (const float*)d_in[18];
    const float* b_ts1  = (const float*)d_in[19];
    const float* W_last = (const float*)d_in[20];
    const float* as_last = (const float*)d_in[21];
    const float* ad_last = (const float*)d_in[22];
    const float* b_last  = (const float*)d_in[23];

    const int Eh = 320000;     // per-direction edge count
    const int Ef = 640000;     // full edge set

    // ---- workspace carve-up (256B aligned slots) ----
    char* ws = (char*)d_ws;
    size_t off = 0;
    auto take = [&](size_t bytes) -> void* {
        void* p = ws + off;
        off += (bytes + 255) & ~(size_t)255;
        return p;
    };
    bf16_t* xb_in = (bf16_t*)take((size_t)NODES * 128 * sizeof(bf16_t));
    bf16_t* xb_l1 = (bf16_t*)take((size_t)NODES * 512 * sizeof(bf16_t));
    bf16_t* xb_l2 = (bf16_t*)take((size_t)NODES * 512 * sizeof(bf16_t));
    bf16_t* Wpk   = (bf16_t*)take((size_t)512 * 256 * sizeof(bf16_t));  // max packed B
    float*  h     = (float*) take((size_t)NODES * 256 * sizeof(float));
    float*  acc   = (float*) take((size_t)NODES * 256 * sizeof(float));
    float*  asrc  = (float*) take((size_t)NODES * 4 * sizeof(float));
    float*  adst  = (float*) take((size_t)NODES * 4 * sizeof(float));
    float*  emax  = (float*) take((size_t)NODES * 4 * sizeof(float));
    float*  esum  = (float*) take((size_t)NODES * 4 * sizeof(float));
    float*  ee    = (float*) take((size_t)(Ef + NODES) * 4 * sizeof(float));

    // ---- input features -> bf16 ----
    int xn = NODES * 128;
    k_f32_to_bf16<<<(xn + 255) / 256, 256, 0, stream>>>(x, xb_in, xn);

    // ---- layer 0 (bidirectional), outputs concat into xb_l1 [N,512] ----
    run_gat_conv(stream, xb_in, 128, 128, W_st0, as_st0, ad_st0, b_st0,
                 st, st + Eh, Eh, 4, 64,
                 Wpk, h, acc, asrc, adst, emax, esum, ee, xb_l1, 512, 0);
    run_gat_conv(stream, xb_in, 128, 128, W_ts0, as_ts0, ad_ts0, b_ts0,
                 ts, ts + Eh, Eh, 4, 64,
                 Wpk, h, acc, asrc, adst, emax, esum, ee, xb_l1, 512, 256);

    // ---- layer 1, outputs concat into xb_l2 [N,512] ----
    run_gat_conv(stream, xb_l1, 512, 512, W_st1, as_st1, ad_st1, b_st1,
                 st, st + Eh, Eh, 4, 64,
                 Wpk, h, acc, asrc, adst, emax, esum, ee, xb_l2, 512, 0);
    run_gat_conv(stream, xb_l1, 512, 512, W_ts1, as_ts1, ad_ts1, b_ts1,
                 ts, ts + Eh, Eh, 4, 64,
                 Wpk, h, acc, asrc, adst, emax, esum, ee, xb_l2, 512, 256);

    // ---- last layer: single head, 40 channels, full edge set ----
    run_gat_conv(stream, xb_l2, 512, 512, W_last, as_last, ad_last, b_last,
                 edge_index, edge_index + Ef, Ef, 1, 40,
                 Wpk, h, acc, asrc, adst, emax, esum, ee, nullptr, 0, 0);

    k_last_softmax<<<(NODES + 127) / 128, 128, 0, stream>>>(acc, b_last, (float*)d_out, NODES, 40);
}
